// LinearBinary_50414326120609
// MI455X (gfx1250) — matrix operand order repaired and checked by reading, not compiled, not run on hardware
//
#include <hip/hip_runtime.h>
#include <hip/hip_bf16.h>

// ---------------------------------------------------------------------------
// LinearBinary: out = (A @ W + b) + ((A@W+b)[:,0] > 0.5 ? +1 : -1)  per row
//   A: [8192, 2048] f32, W: [2048, 2048] f32, b: [2048] f32
//
// Fast path (ws big enough):
//   prep1: A  -> Ah/Al   (bf16 hi/lo split, row-major [m][k])   in d_ws
//   prep2: W  -> Wth/Wtl (bf16 hi/lo split, TRANSPOSED [n][k])  in d_ws
//   k1   : exact fp32 GEMV column 0 -> flags[8192] = +-1        in d_ws
//   k2   : 3xBF16 WMMA GEMM; tiles staged by Tensor Data Mover
//          (tensor_load_to_lds, double-buffered, TDM-padded LDS pitch),
//          epilogue adds bias + flag.
// Fallback path (small ws): flags + fused-conversion GEMM (round-1 design).
// ---------------------------------------------------------------------------

#define BATCH 8192
#define KDIM  2048
#define NDIM  2048

typedef __attribute__((ext_vector_type(16))) __bf16         v16bf;
typedef __attribute__((ext_vector_type(8)))  float          v8f;
typedef __attribute__((ext_vector_type(8)))  unsigned short u16x8;
typedef __attribute__((ext_vector_type(4)))  unsigned short u16x4;
typedef __attribute__((ext_vector_type(4)))  unsigned int   u32x4;
typedef __attribute__((ext_vector_type(8)))  int            i32x8;
typedef __attribute__((ext_vector_type(4)))  int            i32x4;

union FragBF {
    v16bf v;
    u16x8 h8[2];
};

// round-to-nearest-even fp32 -> bf16 bits
static __device__ __forceinline__ unsigned short f2bf(float f) {
    unsigned u = __builtin_bit_cast(unsigned, f);
    u = (u + 0x7fffu + ((u >> 16) & 1u)) >> 16;
    return (unsigned short)u;
}
static __device__ __forceinline__ float bf2f(unsigned short h) {
    return __builtin_bit_cast(float, (unsigned)h << 16);
}

// ---------------------------------------------------------------------------
// TDM: DMA one [128 rows x 32 bf16] tile (row pitch 2048 elements in memory)
// into LDS with 16B padding after every 64B -> LDS row pitch 80B (40 u16),
// matching the WMMA fragment layout and hitting distinct banks.
// D# built per CDNA5 ISA 8.3/8.4 (group0: count/lds/global/type=2;
// group1: data_size=2B, dims, tile dims, stride, pad ctl).
// This toolchain exposes the 6-arg builtin:
//   (u32x4 g0, i32x8 g1, i32x4 g2, i32x4 g3, i32x8 extra, i32 cpol)
// ---------------------------------------------------------------------------
static __device__ __forceinline__ void tdm_load_tile_bf16(
    const unsigned short* gptr, unsigned tensor_dim1, unsigned lds_off) {
    unsigned long long ga = (unsigned long long)(uintptr_t)gptr;
    const unsigned TD0  = 2048u;   // tensor dim0 length (elements)
    const unsigned STR0 = 2048u;   // dim0 stride (elements)

    u32x4 g0;
    g0[0] = 1u;                                   // count=1, user descriptor
    g0[1] = lds_off;                              // lds_addr (bytes)
    g0[2] = (unsigned)(ga & 0xffffffffu);         // global_addr[31:0]
    g0[3] = (unsigned)((ga >> 32) & 0x01ffffffu)  // global_addr[56:32]
            | 0x80000000u;                        // type=2 ("image")

    i32x8 g1;
    // data_size=1 (2B) <<16 | pad_enable<<20 | pad_interval=3 (64B) <<22
    // | pad_amount=3 (4 DWORDs = 16B) <<25 ; workgroup_mask=0
    g1[0] = (int)((1u << 16) | (1u << 20) | (3u << 22) | (3u << 25));
    g1[1] = (int)((TD0 & 0xffffu) << 16);                         // dim0 lo16
    g1[2] = (int)(((TD0 >> 16) & 0xffffu) |
                  ((tensor_dim1 & 0xffffu) << 16));               // dim0 hi | dim1 lo
    g1[3] = (int)(((tensor_dim1 >> 16) & 0xffffu) | (32u << 16)); // dim1 hi | tile_dim0=32
    g1[4] = (int)128u;                                            // tile_dim1=128, tile_dim2=0
    g1[5] = (int)STR0;                                            // dim0_stride lo32
    g1[6] = 0;                                                    // stride hi | dim1_stride lo
    g1[7] = 0;

    i32x4 z4 = {0, 0, 0, 0};                 // groups 2/3 unused (2D tensor)
    i32x8 z8 = {0, 0, 0, 0, 0, 0, 0, 0};     // extra descriptor words (unused)
    __builtin_amdgcn_tensor_load_to_lds(g0, g1, z4, z4, z8, 0);
}

// ---------------------------------------------------------------------------
// prep1: elementwise fp32 -> bf16 hi/lo split (for A). One float4 per thread.
// ---------------------------------------------------------------------------
__global__ void lb_split_kernel(const float* __restrict__ src,
                                unsigned short* __restrict__ dh,
                                unsigned short* __restrict__ dl) {
    size_t idx = (size_t)blockIdx.x * blockDim.x + threadIdx.x;   // float4 index
    float4 v = reinterpret_cast<const float4*>(src)[idx];
    float f[4] = {v.x, v.y, v.z, v.w};
    u16x4 hv, lv;
    #pragma unroll
    for (int i = 0; i < 4; ++i) {
        unsigned short hb = f2bf(f[i]);
        hv[i] = hb;
        lv[i] = f2bf(f[i] - bf2f(hb));
    }
    reinterpret_cast<u16x4*>(dh)[idx] = hv;
    reinterpret_cast<u16x4*>(dl)[idx] = lv;
}

// ---------------------------------------------------------------------------
// prep2: W [k][n] fp32 -> Wt hi/lo bf16 [n][k], 64x64 tiles through LDS.
// ---------------------------------------------------------------------------
__global__ __launch_bounds__(256)
void lb_transpose_split_kernel(const float* __restrict__ W,
                               unsigned short* __restrict__ wth,
                               unsigned short* __restrict__ wtl) {
    __shared__ unsigned short sTh[64 * 72];
    __shared__ unsigned short sTl[64 * 72];
    const int tid   = threadIdx.x;
    const int kBase = blockIdx.y * 64;
    const int nBase = blockIdx.x * 64;

    #pragma unroll
    for (int p = 0; p < 4; ++p) {
        int idx = p * 256 + tid;      // 0..1023
        int k   = idx >> 4;           // 0..63
        int n4  = idx & 15;           // 16 float4 per 64-float row
        float4 v = *reinterpret_cast<const float4*>(
            W + (size_t)(kBase + k) * NDIM + nBase + n4 * 4);
        float f[4] = {v.x, v.y, v.z, v.w};
        #pragma unroll
        for (int i = 0; i < 4; ++i) {
            int n = n4 * 4 + i;
            unsigned short hb = f2bf(f[i]);
            sTh[n * 72 + k] = hb;
            sTl[n * 72 + k] = f2bf(f[i] - bf2f(hb));
        }
    }
    __syncthreads();
    #pragma unroll
    for (int p = 0; p < 4; ++p) {
        int idx = p * 256 + tid;
        int n   = idx >> 4;           // 0..63
        int k4  = idx & 15;
        size_t o = (size_t)(nBase + n) * KDIM + kBase + k4 * 4;
        *reinterpret_cast<u16x4*>(wth + o) =
            *reinterpret_cast<const u16x4*>(&sTh[n * 72 + k4 * 4]);
        *reinterpret_cast<u16x4*>(wtl + o) =
            *reinterpret_cast<const u16x4*>(&sTl[n * 72 + k4 * 4]);
    }
}

// ---------------------------------------------------------------------------
// k1: flags[row] = (dot(A[row,:], W[:,0]) + b[0] > 0.5) ? +1 : -1
// One wave32 per row, exact fp32 FMA.
// ---------------------------------------------------------------------------
__global__ void lb_flags_kernel(const float* __restrict__ A,
                                const float* __restrict__ W,
                                const float* __restrict__ b,
                                float* __restrict__ flags) {
    const int lane = threadIdx.x & 31;
    const int wave = threadIdx.x >> 5;
    const int row  = blockIdx.x * 8 + wave;

    const float* arow = A + (size_t)row * KDIM;
    float s = 0.0f;
    for (int k0 = lane * 4; k0 < KDIM; k0 += 32 * 4) {
        float4 a = *reinterpret_cast<const float4*>(arow + k0);
        s += a.x * W[(size_t)(k0 + 0) * NDIM];
        s += a.y * W[(size_t)(k0 + 1) * NDIM];
        s += a.z * W[(size_t)(k0 + 2) * NDIM];
        s += a.w * W[(size_t)(k0 + 3) * NDIM];
    }
    #pragma unroll
    for (int off = 16; off > 0; off >>= 1)
        s += __shfl_xor(s, off, 32);
    if (lane == 0)
        flags[row] = (s + b[0] > 0.5f) ? 1.0f : -1.0f;
}

// ---------------------------------------------------------------------------
// Shared GEMM geometry
// ---------------------------------------------------------------------------
#define BM 128
#define BN 128
#define BK 32
#define SA 40          // LDS row pitch (u16) = 80B, bank-friendly
#define TILE_U16 (128 * SA)   // 5120 u16 = 10KB per tile

// ---------------------------------------------------------------------------
// k2 (fast): TDM-staged, double-buffered 3xBF16 WMMA GEMM.
// 256 threads = 8 waves (2 M x 4 N), wave tile 64x32 = 4x2 of 16x16.
// Dynamic LDS: 2 buffers x 4 tiles (Ah, Al, Bh, Bl) x 10KB = 80KB.
// ---------------------------------------------------------------------------
__global__ __launch_bounds__(256)
void lb_gemm_tdm_kernel(const unsigned short* __restrict__ Ah,
                        const unsigned short* __restrict__ Al,
                        const unsigned short* __restrict__ Wth,
                        const unsigned short* __restrict__ Wtl,
                        const float* __restrict__ bias,
                        const float* __restrict__ flags,
                        float* __restrict__ out) {
    extern __shared__ unsigned short smem[];   // 8 * TILE_U16

    const int tid  = threadIdx.x;
    const int lane = tid & 31;
    const int wave = tid >> 5;
    const int wm   = wave & 1;
    const int wn   = wave >> 1;
    const int h    = lane >> 4;
    const int m16  = lane & 15;

    const int mBase = blockIdx.y * BM;
    const int nBase = blockIdx.x * BN;

    const unsigned lds0 = (unsigned)(uintptr_t)&smem[0];  // LDS byte offset base

    v8f acc[4][2] = {};

    // prologue: DMA stage 0 into buffer 0
    if (tid == 0) {
        tdm_load_tile_bf16(Ah  + (size_t)mBase * KDIM, BATCH, lds0 + 0 * TILE_U16 * 2);
        tdm_load_tile_bf16(Al  + (size_t)mBase * KDIM, BATCH, lds0 + 1 * TILE_U16 * 2);
        tdm_load_tile_bf16(Wth + (size_t)nBase * KDIM, NDIM,  lds0 + 2 * TILE_U16 * 2);
        tdm_load_tile_bf16(Wtl + (size_t)nBase * KDIM, NDIM,  lds0 + 3 * TILE_U16 * 2);
    }

    for (int it = 0; it < KDIM / BK; ++it) {
        const int kb = it * BK;
        const int p  = it & 1;

        if (tid == 0) {
            if (it + 1 < KDIM / BK) {
                // prefetch next stage into the other buffer, then retire current:
                // TDM completes in issue order -> <=4 outstanding means stage
                // 'it' has fully landed.
                unsigned nb = lds0 + (unsigned)((p ^ 1) * 4) * TILE_U16 * 2;
                tdm_load_tile_bf16(Ah  + (size_t)mBase * KDIM + kb + BK, BATCH, nb + 0 * TILE_U16 * 2);
                tdm_load_tile_bf16(Al  + (size_t)mBase * KDIM + kb + BK, BATCH, nb + 1 * TILE_U16 * 2);
                tdm_load_tile_bf16(Wth + (size_t)nBase * KDIM + kb + BK, NDIM,  nb + 2 * TILE_U16 * 2);
                tdm_load_tile_bf16(Wtl + (size_t)nBase * KDIM + kb + BK, NDIM,  nb + 3 * TILE_U16 * 2);
                __builtin_amdgcn_s_wait_tensorcnt(4);
            } else {
                __builtin_amdgcn_s_wait_tensorcnt(0);
            }
        }
        __syncthreads();   // stage 'it' visible to all waves

        const unsigned short* bAh = smem + (p * 4 + 0) * TILE_U16;
        const unsigned short* bAl = smem + (p * 4 + 1) * TILE_U16;
        const unsigned short* bBh = smem + (p * 4 + 2) * TILE_U16;
        const unsigned short* bBl = smem + (p * 4 + 3) * TILE_U16;

        // B fragments: lane = N (mod 16), K = h*8 + j (j<8), 16 + h*8 + (j-8) (j>=8)
        FragBF fbh[2], fbl[2];
        #pragma unroll
        for (int nt = 0; nt < 2; ++nt) {
            int nb = (wn * 32 + nt * 16 + m16) * SA + h * 8;
            fbh[nt].h8[0] = *reinterpret_cast<const u16x8*>(&bBh[nb]);
            fbh[nt].h8[1] = *reinterpret_cast<const u16x8*>(&bBh[nb + 16]);
            fbl[nt].h8[0] = *reinterpret_cast<const u16x8*>(&bBl[nb]);
            fbl[nt].h8[1] = *reinterpret_cast<const u16x8*>(&bBl[nb + 16]);
        }

        // A fragments + 3-term split WMMA
        #pragma unroll
        for (int mt = 0; mt < 4; ++mt) {
            int ab = (wm * 64 + mt * 16 + m16) * SA;
            FragBF fah, fal;
            fah.h8[0] = *reinterpret_cast<const u16x8*>(&bAh[ab + h * 8]);
            fah.h8[1] = *reinterpret_cast<const u16x8*>(&bAh[ab + 16 + h * 8]);
            fal.h8[0] = *reinterpret_cast<const u16x8*>(&bAl[ab + h * 8]);
            fal.h8[1] = *reinterpret_cast<const u16x8*>(&bAl[ab + 16 + h * 8]);
            #pragma unroll
            for (int nt = 0; nt < 2; ++nt) {
                acc[mt][nt] = __builtin_amdgcn_wmma_f32_16x16x32_bf16(
                    false, fah.v, false, fbh[nt].v, (short)0, acc[mt][nt], false, false);
                acc[mt][nt] = __builtin_amdgcn_wmma_f32_16x16x32_bf16(
                    false, fah.v, false, fbl[nt].v, (short)0, acc[mt][nt], false, false);
                acc[mt][nt] = __builtin_amdgcn_wmma_f32_16x16x32_bf16(
                    false, fal.v, false, fbh[nt].v, (short)0, acc[mt][nt], false, false);
            }
        }
        __syncthreads();   // frag reads done before TDM overwrites this buffer
    }

    // epilogue: + bias[col] + flag[row]
    float bcol[2];
    #pragma unroll
    for (int nt = 0; nt < 2; ++nt)
        bcol[nt] = bias[nBase + wn * 32 + nt * 16 + m16];
    #pragma unroll
    for (int mt = 0; mt < 4; ++mt) {
        #pragma unroll
        for (int r = 0; r < 8; ++r) {
            int row = mBase + wm * 64 + mt * 16 + h * 8 + r;
            float fl = flags[row];
            #pragma unroll
            for (int nt = 0; nt < 2; ++nt) {
                int col = nBase + wn * 32 + nt * 16 + m16;
                out[(size_t)row * NDIM + col] = acc[mt][nt][r] + bcol[nt] + fl;
            }
        }
    }
}

// ---------------------------------------------------------------------------
// k2 (fallback, small ws): fused-conversion GEMM (round-1 design).
// ---------------------------------------------------------------------------
__global__ __launch_bounds__(256)
void lb_gemm_fallback_kernel(const float* __restrict__ A,
                             const float* __restrict__ W,
                             const float* __restrict__ bias,
                             const float* __restrict__ flags,
                             float* __restrict__ out) {
    __shared__ unsigned short sAh[BM * SA];
    __shared__ unsigned short sAl[BM * SA];
    __shared__ unsigned short sBh[BN * SA];
    __shared__ unsigned short sBl[BN * SA];

    const int tid  = threadIdx.x;
    const int lane = tid & 31;
    const int wave = tid >> 5;
    const int wm   = wave & 1;
    const int wn   = wave >> 1;
    const int h    = lane >> 4;
    const int m16  = lane & 15;
    const int mBase = blockIdx.y * BM;
    const int nBase = blockIdx.x * BN;

    v8f acc[4][2] = {};

    for (int kb = 0; kb < KDIM; kb += BK) {
        __syncthreads();
        #pragma unroll
        for (int p = 0; p < 4; ++p) {
            int idx = p * 256 + tid;
            int r   = idx >> 3;
            int c4  = idx & 7;
            float4 v = *reinterpret_cast<const float4*>(
                A + (size_t)(mBase + r) * KDIM + kb + c4 * 4);
            u16x4 hv, lv;
            float f[4] = {v.x, v.y, v.z, v.w};
            #pragma unroll
            for (int i = 0; i < 4; ++i) {
                unsigned short hb = f2bf(f[i]);
                hv[i] = hb;
                lv[i] = f2bf(f[i] - bf2f(hb));
            }
            *reinterpret_cast<u16x4*>(&sAh[r * SA + c4 * 4]) = hv;
            *reinterpret_cast<u16x4*>(&sAl[r * SA + c4 * 4]) = lv;
        }
        #pragma unroll
        for (int p = 0; p < 4; ++p) {
            int idx = p * 256 + tid;
            int k   = idx >> 5;
            int n4  = idx & 31;
            float4 v = *reinterpret_cast<const float4*>(
                W + (size_t)(kb + k) * NDIM + nBase + n4 * 4);
            float f[4] = {v.x, v.y, v.z, v.w};
            #pragma unroll
            for (int i = 0; i < 4; ++i) {
                int n = n4 * 4 + i;
                unsigned short hb = f2bf(f[i]);
                sBh[n * SA + k] = hb;
                sBl[n * SA + k] = f2bf(f[i] - bf2f(hb));
            }
        }
        __syncthreads();

        FragBF fbh[2], fbl[2];
        #pragma unroll
        for (int nt = 0; nt < 2; ++nt) {
            int nb = (wn * 32 + nt * 16 + m16) * SA + h * 8;
            fbh[nt].h8[0] = *reinterpret_cast<const u16x8*>(&sBh[nb]);
            fbh[nt].h8[1] = *reinterpret_cast<const u16x8*>(&sBh[nb + 16]);
            fbl[nt].h8[0] = *reinterpret_cast<const u16x8*>(&sBl[nb]);
            fbl[nt].h8[1] = *reinterpret_cast<const u16x8*>(&sBl[nb + 16]);
        }
        #pragma unroll
        for (int mt = 0; mt < 4; ++mt) {
            int ab = (wm * 64 + mt * 16 + m16) * SA;
            FragBF fah, fal;
            fah.h8[0] = *reinterpret_cast<const u16x8*>(&sAh[ab + h * 8]);
            fah.h8[1] = *reinterpret_cast<const u16x8*>(&sAh[ab + 16 + h * 8]);
            fal.h8[0] = *reinterpret_cast<const u16x8*>(&sAl[ab + h * 8]);
            fal.h8[1] = *reinterpret_cast<const u16x8*>(&sAl[ab + 16 + h * 8]);
            #pragma unroll
            for (int nt = 0; nt < 2; ++nt) {
                acc[mt][nt] = __builtin_amdgcn_wmma_f32_16x16x32_bf16(
                    false, fah.v, false, fbh[nt].v, (short)0, acc[mt][nt], false, false);
                acc[mt][nt] = __builtin_amdgcn_wmma_f32_16x16x32_bf16(
                    false, fah.v, false, fbl[nt].v, (short)0, acc[mt][nt], false, false);
                acc[mt][nt] = __builtin_amdgcn_wmma_f32_16x16x32_bf16(
                    false, fal.v, false, fbh[nt].v, (short)0, acc[mt][nt], false, false);
            }
        }
    }

    float bcol[2];
    #pragma unroll
    for (int nt = 0; nt < 2; ++nt)
        bcol[nt] = bias[nBase + wn * 32 + nt * 16 + m16];
    #pragma unroll
    for (int mt = 0; mt < 4; ++mt) {
        #pragma unroll
        for (int r = 0; r < 8; ++r) {
            int row = mBase + wm * 64 + mt * 16 + h * 8 + r;
            float fl = flags[row];
            #pragma unroll
            for (int nt = 0; nt < 2; ++nt) {
                int col = nBase + wn * 32 + nt * 16 + m16;
                out[(size_t)row * NDIM + col] = acc[mt][nt][r] + bcol[nt] + fl;
            }
        }
    }
}

// ---------------------------------------------------------------------------
extern "C" void kernel_launch(void* const* d_in, const int* in_sizes, int n_in,
                              void* d_out, int out_size, void* d_ws, size_t ws_size,
                              hipStream_t stream) {
    const float* A = (const float*)d_in[0];
    const float* W = (const float*)d_in[1];
    const float* b = (const float*)d_in[2];
    float* out = (float*)d_out;

    // ws layout: [flags 32KB][Ah 32MB][Al 32MB][Wth 8MB][Wtl 8MB]
    char* ws = (char*)d_ws;
    float* flags = (float*)ws;
    const size_t flagsB = (size_t)BATCH * 4;
    const size_t aB     = (size_t)BATCH * KDIM * 2;
    const size_t wB     = (size_t)KDIM * NDIM * 2;
    unsigned short* Ah  = (unsigned short*)(ws + flagsB);
    unsigned short* Al  = (unsigned short*)(ws + flagsB + aB);
    unsigned short* Wth = (unsigned short*)(ws + flagsB + 2 * aB);
    unsigned short* Wtl = (unsigned short*)(ws + flagsB + 2 * aB + wB);
    const size_t need = flagsB + 2 * aB + 2 * wB;

    lb_flags_kernel<<<BATCH / 8, 256, 0, stream>>>(A, W, b, flags);

    dim3 grid(NDIM / BN, BATCH / BM);  // (16, 64)
    if (ws_size >= need) {
        lb_split_kernel<<<(BATCH * KDIM / 4) / 256, 256, 0, stream>>>(A, Ah, Al);
        dim3 tgrid(NDIM / 64, KDIM / 64);  // (32, 32)
        lb_transpose_split_kernel<<<tgrid, 256, 0, stream>>>(W, Wth, Wtl);
        lb_gemm_tdm_kernel<<<grid, 256, 8 * TILE_U16 * 2, stream>>>(
            Ah, Al, Wth, Wtl, b, flags, out);
    } else {
        lb_gemm_fallback_kernel<<<grid, 256, 0, stream>>>(A, W, b, flags, out);
    }
}
